// StructureAwareSSM_15247133901504
// MI455X (gfx1250) — compile-verified, matrix operand order checked by reading, not run on hardware
//
#include <hip/hip_runtime.h>
#include <hip/hip_bf16.h>

// ---------------------------------------------------------------------------
// StructureAwareSSM for MI455X (gfx1250, wave32).
// GEMMs: v_wmma_f32_16x16x32_bf16, operands pre-converted to bf16 (one pass),
// fragments loaded as aligned uint4 (8 bf16) per ISA 7.12.2 layout.
// Each wave: 1 M-tile x 4 N-tiles (shared A fragment, 4 accumulators).
// Scan kernel stages LDS via GLOBAL_LOAD_ASYNC_TO_LDS_B128 (ASYNCcnt).
// Workspace: ~155 MB, fits in the 192 MB L2.
// ---------------------------------------------------------------------------

typedef __attribute__((ext_vector_type(16))) __bf16 v16bf;
typedef __attribute__((ext_vector_type(8)))  float  v8f;

#define LTOT   4096
#define DM     512     // d_model
#define DI     512     // d_inner
#define KDBL   34      // dt_rank + 2*d_state
#define RRANK  32

// LDS byte offset of a __shared__ object (generic -> addrspace(3) cast).
__device__ __forceinline__ unsigned lds_addr_of(const void* p) {
  return (unsigned)(unsigned long long)(__attribute__((address_space(3))) const void*)p;
}

// ---- fp32 -> packed bf16 (RNE), one dword = 2 elements --------------------
__global__ void __launch_bounds__(256)
k_f2bf(const float* __restrict__ in, unsigned int* __restrict__ out, int n2) {
  int i = blockIdx.x * 256 + threadIdx.x;
  if (i >= n2) return;
  float2 f = ((const float2*)in)[i];
  unsigned int u0 = __builtin_bit_cast(unsigned int, f.x);
  unsigned int u1 = __builtin_bit_cast(unsigned int, f.y);
  unsigned int r0 = (u0 + 0x7FFFu + ((u0 >> 16) & 1u)) >> 16;
  unsigned int r1 = (u1 + 0x7FFFu + ((u1 >> 16) & 1u)) & 0xFFFF0000u;
  out[i] = r1 | r0;
}

// Load one 16x32 bf16 fragment half-pair: elements 0..7 <- K o0..o0+7,
// elements 8..15 <- K o1..o1+7 (ISA 7.12.2 16-bit A/B layout).
__device__ __forceinline__ v16bf load_frag(const unsigned short* rowp, int o0, int o1) {
  union { v16bf v; uint4 q[2]; } u;
  u.q[0] = *(const uint4*)(rowp + o0);
  u.q[1] = *(const uint4*)(rowp + o1);
  return u.v;
}

// ---------------------------------------------------------------------------
// GEMM1: xz[m,n] = sum_k X[m,k]*in_proj_w[n,k]; M=16384,K=512,N=1024 (bf16).
// Wave = 16 rows x 64 cols (4 tiles, shared A).  grid=(4,1024), block=128.
// n<512 -> xin channel-major (b,n,l); n>=512 -> z row-major (m,n-512).
// ---------------------------------------------------------------------------
__global__ void __launch_bounds__(128)
k_gemm_inproj(const unsigned short* __restrict__ Xbf,
              const unsigned short* __restrict__ Wbf,
              float* __restrict__ xin, float* __restrict__ zbuf) {
  const int lane   = threadIdx.x & 31;
  const int wave   = threadIdx.x >> 5;
  const int laneLo = lane & 15;
  const int hi     = lane >> 4;
  const int nTile0 = (blockIdx.x * 4 + wave) * 4;     // 4 consecutive N tiles
  const int mTile  = blockIdx.y;
  const int row    = mTile * 16 + laneLo;
  const unsigned short* arow = Xbf + (size_t)row * DM;
  const unsigned short* brow0 = Wbf + (size_t)((nTile0 + 0) * 16 + laneLo) * DM;
  const unsigned short* brow1 = Wbf + (size_t)((nTile0 + 1) * 16 + laneLo) * DM;
  const unsigned short* brow2 = Wbf + (size_t)((nTile0 + 2) * 16 + laneLo) * DM;
  const unsigned short* brow3 = Wbf + (size_t)((nTile0 + 3) * 16 + laneLo) * DM;

  v8f acc0 = {}, acc1 = {}, acc2 = {}, acc3 = {};
#pragma unroll
  for (int kk = 0; kk < DM / 32; ++kk) {
    const int k0 = kk * 32;
    const int o0 = k0 + hi * 8;
    const int o1 = k0 + 16 + hi * 8;
    if (kk + 1 < DM / 32) {
      __builtin_prefetch(arow + k0 + 32, 0, 3);
      __builtin_prefetch(brow0 + k0 + 32, 0, 3);
    }
    v16bf va = load_frag(arow, o0, o1);
    v16bf b0 = load_frag(brow0, o0, o1);
    v16bf b1 = load_frag(brow1, o0, o1);
    v16bf b2 = load_frag(brow2, o0, o1);
    v16bf b3 = load_frag(brow3, o0, o1);
    acc0 = __builtin_amdgcn_wmma_f32_16x16x32_bf16(false, va, false, b0, (short)0, acc0, false, false);
    acc1 = __builtin_amdgcn_wmma_f32_16x16x32_bf16(false, va, false, b1, (short)0, acc1, false, false);
    acc2 = __builtin_amdgcn_wmma_f32_16x16x32_bf16(false, va, false, b2, (short)0, acc2, false, false);
    acc3 = __builtin_amdgcn_wmma_f32_16x16x32_bf16(false, va, false, b3, (short)0, acc3, false, false);
  }
  v8f accs[4] = {acc0, acc1, acc2, acc3};
#pragma unroll
  for (int t = 0; t < 4; ++t) {
    const int nc = (nTile0 + t) * 16 + laneLo;
#pragma unroll
    for (int i = 0; i < 8; ++i) {       // C/D layout: VGPR i -> M = i + 8*hi
      int m = mTile * 16 + hi * 8 + i;
      int b = m >> 12, l = m & 4095;
      float v = accs[t][i];
      if (nc < DI) xin[((size_t)(b * DI + nc)) * LTOT + l] = v;
      else         zbuf[(size_t)m * DI + (nc - DI)] = v;
    }
  }
}

// ---------------------------------------------------------------------------
// Depthwise 3x3 conv (zero-pad SAME) + bias + SiLU.  xin NCHW -> xs (b,c,l).
// ---------------------------------------------------------------------------
__global__ void __launch_bounds__(256)
k_dwconv_silu(const float* __restrict__ xin, const float* __restrict__ w,
              const float* __restrict__ bias, float* __restrict__ xs) {
  int idx = blockIdx.x * 256 + threadIdx.x;       // 4*512*4096 threads
  int l = idx & 4095;
  int c = (idx >> 12) & 511;
  int b = idx >> 21;
  int y = l >> 6, x = l & 63;
  const float* base = xin + ((size_t)(b * DI + c)) * LTOT;
  const float* wc = w + c * 9;
  float acc = bias[c];
#pragma unroll
  for (int i = 0; i < 3; ++i) {
    int yy = y + i - 1;
    if (yy < 0 || yy > 63) continue;
#pragma unroll
    for (int j = 0; j < 3; ++j) {
      int xx = x + j - 1;
      if (xx < 0 || xx > 63) continue;
      acc += wc[i * 3 + j] * base[yy * 64 + xx];
    }
  }
  xs[idx] = acc / (1.0f + expf(-acc));            // silu
}

// x_dbl[b,k,l] = sum_d x_proj_weight[k,d] * xs[b,d,l]
__global__ void __launch_bounds__(256)
k_xdbl(const float* __restrict__ xs, const float* __restrict__ xpw,
       float* __restrict__ xdbl) {
  int l = blockIdx.x * 256 + threadIdx.x;
  int k = blockIdx.y;
  int b = blockIdx.z;
  const float* col = xs + (size_t)b * DI * LTOT + l;
  const float* wr  = xpw + k * DI;
  float acc = 0.f;
#pragma unroll 8
  for (int d = 0; d < DI; ++d) acc += wr[d] * col[(size_t)d * LTOT];
  xdbl[((size_t)(b * KDBL + k)) * LTOT + l] = acc;
}

// dts[b,d,l] = sum_r dt_projs_weight[d,r] * x_dbl[b,r,l]
__global__ void __launch_bounds__(256)
k_dts(const float* __restrict__ xdbl, const float* __restrict__ dtw,
      float* __restrict__ dts) {
  int l = blockIdx.x * 256 + threadIdx.x;
  int d = blockIdx.y;
  int b = blockIdx.z;
  const float* wr   = dtw + d * RRANK;
  const float* base = xdbl + (size_t)b * KDBL * LTOT + l;
  float acc = 0.f;
#pragma unroll
  for (int r = 0; r < RRANK; ++r) acc += wr[r] * base[(size_t)r * LTOT];
  dts[((size_t)(b * DI + d)) * LTOT + l] = acc;
}

// ---------------------------------------------------------------------------
// Selective-scan: per (b,d) sequence of 4096, h[t] = a[t]*h[t-1] + bb[t].
// 128 threads x 32 contiguous elements.  The 48KB of inputs are staged into
// LDS with GLOBAL_LOAD_ASYNC_TO_LDS_B128 (no VGPR round-trip, ASYNCcnt),
// then a block-wide inclusive scan of (A,B) affine pairs (Hillis-Steele).
// ---------------------------------------------------------------------------
__global__ void __launch_bounds__(128)
k_scan(const float* __restrict__ dts, const float* __restrict__ xs,
       const float* __restrict__ xdbl, const float* __restrict__ dt_bias,
       const float* __restrict__ A_logs, float* __restrict__ hbuf) {
  __shared__ float s_dts[LTOT];
  __shared__ float s_xs [LTOT];
  __shared__ float s_bs [LTOT];
  __shared__ float sA[128];
  __shared__ float sB[128];
  const int tid = threadIdx.x;
  const int d = blockIdx.x;
  const int b = blockIdx.y;
  const size_t rowDL = ((size_t)(b * DI + d)) * LTOT;
  const size_t rowBs = ((size_t)(b * KDBL + RRANK)) * LTOT;   // Bs = row 32

  // async global -> LDS copy: 3 rows x 16KB, 128 lanes x 16B per step
  {
    const unsigned long long gdts = (unsigned long long)(dts + rowDL);
    const unsigned long long gxs  = (unsigned long long)(xs + rowDL);
    const unsigned long long gbs  = (unsigned long long)(xdbl + rowBs);
    const unsigned l_dts = lds_addr_of(s_dts);
    const unsigned l_xs  = lds_addr_of(s_xs);
    const unsigned l_bs  = lds_addr_of(s_bs);
#pragma unroll
    for (int j = 0; j < 8; ++j) {
      const unsigned boff = (unsigned)(j * 128 + tid) * 16u;   // byte offset
      asm volatile("global_load_async_to_lds_b128 %0, %1, off"
                   :: "v"(l_dts + boff), "v"(gdts + boff) : "memory");
      asm volatile("global_load_async_to_lds_b128 %0, %1, off"
                   :: "v"(l_xs + boff), "v"(gxs + boff) : "memory");
      asm volatile("global_load_async_to_lds_b128 %0, %1, off"
                   :: "v"(l_bs + boff), "v"(gbs + boff) : "memory");
    }
  }
  const float Ad    = -expf(A_logs[d]);     // overlaps with async copy
  const float dbias = dt_bias[d];
  asm volatile("s_wait_asynccnt 0" ::: "memory");
  __syncthreads();

  float aarr[32], barr[32];
  float segA = 1.f, segB = 0.f;
  const int base = tid * 32;
#pragma unroll
  for (int j = 0; j < 32; ++j) {
    float xv = s_dts[base + j] + dbias;
    float delta = (xv > 20.f) ? xv : log1pf(expf(xv));   // softplus
    float a  = expf(delta * Ad);
    float bb = delta * s_bs[base + j] * s_xs[base + j];
    aarr[j] = a; barr[j] = bb;
    segA = a * segA;
    segB = a * segB + bb;
  }
  sA[tid] = segA; sB[tid] = segB;
  __syncthreads();
  for (int off = 1; off < 128; off <<= 1) {
    float lA = 1.f, lB = 0.f;
    if (tid >= off) { lA = sA[tid - off]; lB = sB[tid - off]; }
    float cA = sA[tid], cB = sB[tid];
    __syncthreads();
    sA[tid] = cA * lA;            // compose earlier segment then current
    sB[tid] = cA * lB + cB;
    __syncthreads();
  }
  float h = (tid == 0) ? 0.f : sB[tid - 1];   // exclusive prefix (h0 = 0)
#pragma unroll
  for (int j = 0; j < 32; ++j) {
    h = aarr[j] * h + barr[j];
    hbuf[rowDL + base + j] = h;
  }
}

// ---------------------------------------------------------------------------
// Three dilated (1,2,3) edge-clamped depthwise 3x3 convs, alpha-combined,
// then y = h*Cs + xs*Ds, written row-major (b,l,c) for LayerNorm.
// ---------------------------------------------------------------------------
__global__ void __launch_bounds__(256)
k_sfconv(const float* __restrict__ hbuf, const float* __restrict__ xs,
         const float* __restrict__ xdbl, const float* __restrict__ Ds,
         const float* __restrict__ w1, const float* __restrict__ w2,
         const float* __restrict__ w3, const float* __restrict__ alpha,
         float* __restrict__ yraw) {
  int idx = blockIdx.x * 256 + threadIdx.x;
  int l = idx & 4095;
  int c = (idx >> 12) & 511;
  int b = idx >> 21;
  int y = l >> 6, x = l & 63;
  const float* base = hbuf + ((size_t)(b * DI + c)) * LTOT;
  const float* wk[3] = { w1 + c * 9, w2 + c * 9, w3 + c * 9 };
  float accs[3] = {0.f, 0.f, 0.f};
#pragma unroll
  for (int k = 0; k < 3; ++k) {
    const int dil = k + 1;
#pragma unroll
    for (int i = 0; i < 3; ++i) {
      int yy = y + (i - 1) * dil; yy = yy < 0 ? 0 : (yy > 63 ? 63 : yy);
#pragma unroll
      for (int j = 0; j < 3; ++j) {
        int xx = x + (j - 1) * dil; xx = xx < 0 ? 0 : (xx > 63 ? 63 : xx);
        accs[k] += wk[k][i * 3 + j] * base[yy * 64 + xx];
      }
    }
  }
  float hv = alpha[0] * accs[0] + alpha[1] * accs[1] + alpha[2] * accs[2];
  float Cs = xdbl[((size_t)(b * KDBL + RRANK + 1)) * LTOT + l];  // row 33
  float yv = hv * Cs + xs[idx] * Ds[c];
  yraw[((size_t)(b * LTOT + l)) * DI + c] = yv;
}

// LayerNorm over C=512 per row, * silu(z), emit bf16 A-matrix for out_proj.
__global__ void __launch_bounds__(256)
k_ln_gate(const float* __restrict__ yraw, const float* __restrict__ zbuf,
          const float* __restrict__ lnw, const float* __restrict__ lnb,
          unsigned short* __restrict__ ybf) {
  __shared__ float rs[256], rq[256];
  const int tid = threadIdx.x;
  const size_t row = blockIdx.x;
  const float v0 = yraw[row * DI + tid];
  const float v1 = yraw[row * DI + tid + 256];
  rs[tid] = v0 + v1;
  rq[tid] = v0 * v0 + v1 * v1;
  __syncthreads();
  for (int off = 128; off > 0; off >>= 1) {
    if (tid < off) { rs[tid] += rs[tid + off]; rq[tid] += rq[tid + off]; }
    __syncthreads();
  }
  const float mean = rs[0] * (1.f / 512.f);
  const float var  = rq[0] * (1.f / 512.f) - mean * mean;
  const float rinv = rsqrtf(var + 1e-5f);
#pragma unroll
  for (int half = 0; half < 2; ++half) {
    int c = tid + half * 256;
    float v = half ? v1 : v0;
    float yn = (v - mean) * rinv * lnw[c] + lnb[c];
    float z  = zbuf[row * DI + c];
    float o  = yn * (z / (1.f + expf(-z)));
    unsigned int u = __builtin_bit_cast(unsigned int, o);
    ybf[row * DI + c] = (unsigned short)((u + 0x7FFFu + ((u >> 16) & 1u)) >> 16);
  }
}

// ---------------------------------------------------------------------------
// GEMM2: out[m,j] = sum_d y[m,d]*out_proj_w[j,d]; M=16384,K=512,N=512 (bf16).
// Wave = 16 rows x 64 cols (4 tiles, shared A). grid=(2,1024), block=128.
// ---------------------------------------------------------------------------
__global__ void __launch_bounds__(128)
k_gemm_out(const unsigned short* __restrict__ Ybf,
           const unsigned short* __restrict__ Wbf,
           float* __restrict__ out) {
  const int lane   = threadIdx.x & 31;
  const int wave   = threadIdx.x >> 5;
  const int laneLo = lane & 15;
  const int hi     = lane >> 4;
  const int nTile0 = (blockIdx.x * 4 + wave) * 4;
  const int mTile  = blockIdx.y;
  const int row    = mTile * 16 + laneLo;
  const unsigned short* arow = Ybf + (size_t)row * DI;
  const unsigned short* brow0 = Wbf + (size_t)((nTile0 + 0) * 16 + laneLo) * DI;
  const unsigned short* brow1 = Wbf + (size_t)((nTile0 + 1) * 16 + laneLo) * DI;
  const unsigned short* brow2 = Wbf + (size_t)((nTile0 + 2) * 16 + laneLo) * DI;
  const unsigned short* brow3 = Wbf + (size_t)((nTile0 + 3) * 16 + laneLo) * DI;

  v8f acc0 = {}, acc1 = {}, acc2 = {}, acc3 = {};
#pragma unroll
  for (int kk = 0; kk < DI / 32; ++kk) {
    const int k0 = kk * 32;
    const int o0 = k0 + hi * 8;
    const int o1 = k0 + 16 + hi * 8;
    if (kk + 1 < DI / 32) {
      __builtin_prefetch(arow + k0 + 32, 0, 3);
      __builtin_prefetch(brow0 + k0 + 32, 0, 3);
    }
    v16bf va = load_frag(arow, o0, o1);
    v16bf b0 = load_frag(brow0, o0, o1);
    v16bf b1 = load_frag(brow1, o0, o1);
    v16bf b2 = load_frag(brow2, o0, o1);
    v16bf b3 = load_frag(brow3, o0, o1);
    acc0 = __builtin_amdgcn_wmma_f32_16x16x32_bf16(false, va, false, b0, (short)0, acc0, false, false);
    acc1 = __builtin_amdgcn_wmma_f32_16x16x32_bf16(false, va, false, b1, (short)0, acc1, false, false);
    acc2 = __builtin_amdgcn_wmma_f32_16x16x32_bf16(false, va, false, b2, (short)0, acc2, false, false);
    acc3 = __builtin_amdgcn_wmma_f32_16x16x32_bf16(false, va, false, b3, (short)0, acc3, false, false);
  }
  v8f accs[4] = {acc0, acc1, acc2, acc3};
#pragma unroll
  for (int t = 0; t < 4; ++t) {
    const int nc = (nTile0 + t) * 16 + laneLo;
#pragma unroll
    for (int i = 0; i < 8; ++i) {
      int m = mTile * 16 + hi * 8 + i;
      out[(size_t)m * DM + nc] = accs[t][i];
    }
  }
}

// ---------------------------------------------------------------------------
extern "C" void kernel_launch(void* const* d_in, const int* in_sizes, int n_in,
                              void* d_out, int out_size, void* d_ws, size_t ws_size,
                              hipStream_t stream) {
  (void)in_sizes; (void)n_in; (void)out_size; (void)ws_size;
  const float* x      = (const float*)d_in[0];
  const float* ipw    = (const float*)d_in[1];
  const float* convw  = (const float*)d_in[2];
  const float* convb  = (const float*)d_in[3];
  const float* xpw    = (const float*)d_in[4];
  const float* dtw    = (const float*)d_in[5];
  const float* dtb    = (const float*)d_in[6];
  const float* A_logs = (const float*)d_in[7];
  const float* Ds     = (const float*)d_in[8];
  const float* k1     = (const float*)d_in[9];
  const float* k2     = (const float*)d_in[10];
  const float* k3     = (const float*)d_in[11];
  const float* alpha  = (const float*)d_in[12];
  const float* lnw    = (const float*)d_in[13];
  const float* lnb    = (const float*)d_in[14];
  const float* opw    = (const float*)d_in[15];
  float* out = (float*)d_out;

  const size_t SZ = (size_t)4 * DI * LTOT;            // 8,388,608 elements
  float* xin  = (float*)d_ws;                         // (b,c,l); reused as h
  float* zbuf = xin + SZ;                             // (m,c)
  float* xs   = zbuf + SZ;                            // (b,c,l)
  float* dts  = xs + SZ;                              // (b,d,l); reused as yraw
  float* xdbl = dts + SZ;                             // (b,34,l)
  unsigned short* bfA  = (unsigned short*)(xdbl + (size_t)4 * KDBL * LTOT);
  //   bfA: bf16 X [16384,512] for GEMM1, later reused as bf16 Y for GEMM2
  unsigned short* bfW  = bfA + SZ;                    // bf16 in_proj_w [1024,512]
  unsigned short* bfW2 = bfW + (size_t)1024 * DM;     // bf16 out_proj_w [512,512]

  // one-pass bf16 conversions (2 elements per thread)
  k_f2bf<<<(int)(SZ / 512), 256, 0, stream>>>(x,   (unsigned int*)bfA,  (int)(SZ / 2));
  k_f2bf<<<1024, 256, 0, stream>>>(ipw, (unsigned int*)bfW,  1024 * DM / 2);
  k_f2bf<<<512, 256, 0, stream>>>(opw, (unsigned int*)bfW2, DI * DM / 2);

  k_gemm_inproj<<<dim3(4, 1024), 128, 0, stream>>>(bfA, bfW, xin, zbuf);
  k_dwconv_silu<<<(int)(SZ / 256), 256, 0, stream>>>(xin, convw, convb, xs);
  k_xdbl<<<dim3(16, KDBL, 4), 256, 0, stream>>>(xs, xpw, xdbl);
  k_dts<<<dim3(16, DI, 4), 256, 0, stream>>>(xdbl, dtw, dts);
  k_scan<<<dim3(DI, 4), 128, 0, stream>>>(dts, xs, xdbl, dtb, A_logs, /*h=*/xin);
  k_sfconv<<<(int)(SZ / 256), 256, 0, stream>>>(xin, xs, xdbl, Ds, k1, k2, k3,
                                                alpha, /*yraw=*/dts);
  k_ln_gate<<<16384, 256, 0, stream>>>(dts, zbuf, lnw, lnb, /*ybf=*/bfA);
  k_gemm_out<<<dim3(2, 1024), 128, 0, stream>>>(bfA, bfW2, out);
}
